// LSTM_7078106104542
// MI455X (gfx1250) — matrix operand order repaired and checked by reading, not compiled, not run on hardware
//
#include <hip/hip_runtime.h>
#include <hip/hip_bf16.h>
#include <math.h>

// ---------------------------------------------------------------------------
// LSTM LM forward for MI455X (gfx1250): bf16 WMMA GEMMs, f32 accumulate.
// ---------------------------------------------------------------------------

typedef __bf16 bf16;
typedef __bf16 bf16x8  __attribute__((ext_vector_type(8)));
typedef __bf16 bf16x16 __attribute__((ext_vector_type(16)));
typedef float  v8f     __attribute__((ext_vector_type(8)));

#define NVOC   32000
#define NINPUT 512
#define NHID   512
#define TSTEPS 128
#define BATCH  32

union Frag { bf16x16 v; bf16x8 h[2]; };

__device__ __forceinline__ bf16x8 load8(const bf16* p) {
    return *(const bf16x8*)p;
}

// ---------------------------------------------------------------------------
// Generic WMMA GEMM:  C[M,N](f32) = A[M,K](bf16 row-major) * B[K,N](bf16 row-
// major, N-contiguous) [+ bias[n]] [+ addm[m*ldc+n]].
// One wave computes a 32(m) x 64(n) tile: 8 accumulators, 2 A fragments
// shared across 4 B fragments per 32-wide K step.
// Requires M%32==0, N%64==0, K%32==0.
// NT: use non-temporal stores for C (write-once outputs; keeps L2 for B).
// ---------------------------------------------------------------------------
template <bool NT>
__global__ void gemm_bf16_wmma(const bf16* __restrict__ A,
                               const bf16* __restrict__ B,
                               const float* __restrict__ bias,
                               const float* __restrict__ addm,
                               float* __restrict__ C,
                               int M, int N, int K, int ldc)
{
    const int lane = threadIdx.x & 31;
    const int wave = (int)((blockIdx.x * blockDim.x + threadIdx.x) >> 5);
    const int tiles_n = N >> 6;
    const int tiles_m = M >> 5;
    if (wave >= tiles_m * tiles_n) return;
    const int tm = wave / tiles_n;
    const int tn = wave - tm * tiles_n;
    const int m0 = tm << 5;
    const int n0 = tn << 6;

    // A fragment addressing (ISA 7.12.2, 16-bit A 16x32):
    //   lane L: row m = m0 + (L&15); K chunks at ks=(L>>4)*8 and 16+ks.
    const int ks = (lane >> 4) << 3;
    const bf16* pa0 = A + (size_t)(m0 + (lane & 15)) * K + ks;
    const bf16* pa1 = pa0 + (size_t)16 * K;
    // B fragment addressing (dense 32x16 bf16 B): lane L holds column
    // N=n0+(L&15), K at k0+ks+(0..7) and k0+16+ks+(0..7).
    const bf16* pb = B + (size_t)ks * N + n0 + (lane & 15);

    v8f acc[8];
#pragma unroll
    for (int i = 0; i < 8; ++i) acc[i] = (v8f){};

    for (int k0 = 0; k0 < K; k0 += 32) {
        Frag a0, a1;
        a0.h[0] = load8(pa0 + k0);
        a0.h[1] = load8(pa0 + k0 + 16);
        a1.h[0] = load8(pa1 + k0);
        a1.h[1] = load8(pa1 + k0 + 16);
        const bf16* pbk = pb + (size_t)k0 * N;
#pragma unroll
        for (int nt = 0; nt < 4; ++nt) {
            Frag bt;
            for (int s = 0; s < 8; ++s) {
                bt.v[s]     = pbk[(size_t)s * N + nt * 16];
                bt.v[8 + s] = pbk[(size_t)(16 + s) * N + nt * 16];
            }
            acc[nt] = __builtin_amdgcn_wmma_f32_16x16x32_bf16(
                false, a0.v, false, bt.v, (short)0, acc[nt], false, false);
            acc[4 + nt] = __builtin_amdgcn_wmma_f32_16x16x32_bf16(
                false, a1.v, false, bt.v, (short)0, acc[4 + nt], false, false);
        }
    }

    // Epilogue. C layout (32-bit C/D 16x16): VGPR r, lane L -> M=r+8*(L>>4),
    // N = L&15.
    const int nn    = lane & 15;
    const int mbase = m0 + ((lane >> 4) << 3);
#pragma unroll
    for (int nt = 0; nt < 4; ++nt) {
        const int n = n0 + nt * 16 + nn;
        const float bv = bias ? bias[n] : 0.0f;
#pragma unroll
        for (int mt = 0; mt < 2; ++mt) {
            const int mrow = mbase + mt * 16;
#pragma unroll
            for (int r = 0; r < 8; ++r) {
                const size_t idx = (size_t)(mrow + r) * (size_t)ldc + (size_t)n;
                float v = acc[mt * 4 + nt][r] + bv;
                if (addm) v += addm[idx];
                if (NT) __builtin_nontemporal_store(v, &C[idx]);
                else    C[idx] = v;
            }
        }
    }
}

// ---------------------------------------------------------------------------
// Embedding gather + f32 -> bf16.
// ---------------------------------------------------------------------------
__global__ void gather_embed(const int* __restrict__ ids,
                             const float* __restrict__ embed_w,
                             bf16* __restrict__ emb, int total)
{
    int i = blockIdx.x * blockDim.x + threadIdx.x;
    if (i >= total) return;
    int row = i >> 9;            // NINPUT == 512
    int k   = i & 511;
    emb[i] = (bf16)embed_w[(size_t)ids[row] * NINPUT + k];
}

__global__ void f32_to_bf16(const float* __restrict__ in,
                            bf16* __restrict__ out, int n)
{
    int i = blockIdx.x * blockDim.x + threadIdx.x;
    if (i < n) out[i] = (bf16)in[i];
}

// in[R][Cc] (f32) -> out[Cc][R] (bf16), LDS-tiled.  R%32==0, Cc%32==0.
__global__ void transpose_f32_to_bf16(const float* __restrict__ in,
                                      bf16* __restrict__ out, int R, int Cc)
{
    __shared__ float tile[32][33];
    const int c0 = blockIdx.x * 32;
    const int r0 = blockIdx.y * 32;
    const int tx = threadIdx.x;
    const int ty = threadIdx.y;
    for (int i = ty; i < 32; i += 8)
        tile[i][tx] = in[(size_t)(r0 + i) * Cc + (c0 + tx)];
    __syncthreads();
    for (int i = ty; i < 32; i += 8)
        out[(size_t)(c0 + i) * R + (r0 + tx)] = (bf16)tile[tx][i];
}

// ---------------------------------------------------------------------------
// LSTM state init / pointwise update / final copy.
// ---------------------------------------------------------------------------
__global__ void zero_state(float* __restrict__ c, float* __restrict__ h,
                           bf16* __restrict__ h_bf)
{
    int i = blockIdx.x * blockDim.x + threadIdx.x;
    if (i >= BATCH * NHID) return;
    c[i] = 0.0f; h[i] = 0.0f; h_bf[i] = (bf16)0.0f;
}

__global__ void lstm_pointwise(const float* __restrict__ gates,  // [B, 4*NHID]
                               float* __restrict__ c,            // [B*NHID]
                               float* __restrict__ h,            // [B*NHID]
                               bf16* __restrict__ h_bf,          // [B*NHID]
                               bf16* __restrict__ out_bf_t)      // outputs + t*B*NHID
{
    int j = blockIdx.x * blockDim.x + threadIdx.x;
    if (j >= BATCH * NHID) return;
    int b = j >> 9;
    int u = j & 511;
    const float* g = gates + (size_t)b * (4 * NHID);
    float i_ = 1.0f / (1.0f + expf(-g[u]));
    float f_ = 1.0f / (1.0f + expf(-g[NHID + u]));
    float g_ = tanhf(g[2 * NHID + u]);
    float o_ = 1.0f / (1.0f + expf(-g[3 * NHID + u]));
    float cN = f_ * c[j] + i_ * g_;
    float hN = o_ * tanhf(cN);
    c[j] = cN;
    h[j] = hN;
    bf16 hb = (bf16)hN;
    h_bf[j]     = hb;
    out_bf_t[j] = hb;
}

__global__ void copy_hc(const float* __restrict__ h,
                        const float* __restrict__ c,
                        float* __restrict__ out)
{
    int i = blockIdx.x * blockDim.x + threadIdx.x;
    const int n = BATCH * NHID;
    if (i < n)           out[i] = h[i];
    else if (i < 2 * n)  out[i] = c[i - n];
}

// ---------------------------------------------------------------------------
// Launcher.
// ---------------------------------------------------------------------------
static inline int cdiv(int a, int b) { return (a + b - 1) / b; }

extern "C" void kernel_launch(void* const* d_in, const int* in_sizes, int n_in,
                              void* d_out, int out_size, void* d_ws, size_t ws_size,
                              hipStream_t stream)
{
    (void)in_sizes; (void)n_in; (void)out_size; (void)ws_size;

    const int*   ids     = (const int*)  d_in[0];   // [128, 32]
    const float* embed_w = (const float*)d_in[1];   // [32000, 512]
    const float* W       = (const float*)d_in[2];   // [512, 2048]
    const float* U       = (const float*)d_in[3];   // [512, 2048]
    const float* b       = (const float*)d_in[4];   // [2048]
    const float* dec_w   = (const float*)d_in[5];   // [32000, 512]
    const float* dec_b   = (const float*)d_in[6];   // [32000]
    float*       out     = (float*)d_out;

    const int M  = TSTEPS * BATCH;                  // 4096
    const int G  = 4 * NHID;                        // 2048

    // Workspace carve-up (all sizes multiples of 256 bytes).
    char* ws = (char*)d_ws;
    size_t off = 0;
    bf16*  emb_bf  = (bf16*)(ws + off);  off += (size_t)M * NINPUT * 2;       // 4 MB
    bf16*  W_bf    = (bf16*)(ws + off);  off += (size_t)NINPUT * G * 2;       // 2 MB
    bf16*  U_bf    = (bf16*)(ws + off);  off += (size_t)NHID * G * 2;         // 2 MB
    bf16*  decT_bf = (bf16*)(ws + off);  off += (size_t)NHID * NVOC * 2;      // 32 MB
    float* xW      = (float*)(ws + off); off += (size_t)M * G * 4;            // 33.5 MB
    float* gates   = (float*)(ws + off); off += (size_t)BATCH * G * 4;        // 256 KB
    bf16*  h_bf    = (bf16*)(ws + off);  off += (size_t)BATCH * NHID * 2;     // 32 KB
    bf16*  out_bf  = (bf16*)(ws + off);  off += (size_t)M * NHID * 2;         // 4 MB
    float* h_f     = (float*)(ws + off); off += (size_t)BATCH * NHID * 4;     // 64 KB
    float* c_f     = (float*)(ws + off); off += (size_t)BATCH * NHID * 4;     // 64 KB

    // --- state init --------------------------------------------------------
    zero_state<<<cdiv(BATCH * NHID, 256), 256, 0, stream>>>(c_f, h_f, h_bf);

    // --- operand prep ------------------------------------------------------
    gather_embed<<<cdiv(M * NINPUT, 256), 256, 0, stream>>>(ids, embed_w, emb_bf,
                                                            M * NINPUT);
    f32_to_bf16<<<cdiv(NINPUT * G, 256), 256, 0, stream>>>(W, W_bf, NINPUT * G);
    f32_to_bf16<<<cdiv(NHID * G, 256), 256, 0, stream>>>(U, U_bf, NHID * G);
    transpose_f32_to_bf16<<<dim3(NHID / 32, NVOC / 32), dim3(32, 8), 0, stream>>>(
        dec_w, decT_bf, NVOC, NHID);

    // --- xW = emb @ W + b  (M=4096, N=2048, K=512) -------------------------
    {
        int waves = (M / 32) * (G / 64);            // 4096
        gemm_bf16_wmma<false><<<cdiv(waves * 32, 128), 128, 0, stream>>>(
            emb_bf, W_bf, b, nullptr, xW, M, G, NINPUT, G);
    }

    // --- recurrence: 128 sequential steps ----------------------------------
    {
        int waves = (BATCH / 32) * (G / 64);        // 32
        int gblk  = cdiv(waves * 32, 128);
        int pblk  = cdiv(BATCH * NHID, 256);
        for (int t = 0; t < TSTEPS; ++t) {
            gemm_bf16_wmma<false><<<gblk, 128, 0, stream>>>(
                h_bf, U_bf, nullptr, xW + (size_t)t * BATCH * G,
                gates, BATCH, G, NHID, G);
            lstm_pointwise<<<pblk, 256, 0, stream>>>(
                gates, c_f, h_f, h_bf, out_bf + (size_t)t * BATCH * NHID);
        }
    }

    // --- decoder: out = outputs @ dec_w^T + dec_b  (M=4096, N=32000, K=512)
    // Non-temporal stores: the 524 MB output is write-once; keep L2 for B.
    {
        int waves = (M / 32) * (NVOC / 64);         // 64000
        gemm_bf16_wmma<true><<<cdiv(waves * 32, 128), 128, 0, stream>>>(
            out_bf, decT_bf, dec_b, nullptr, out, M, NVOC, NHID, NVOC);
    }

    // --- trailing (h_t, c_t) -----------------------------------------------
    copy_hc<<<cdiv(2 * BATCH * NHID, 256), 256, 0, stream>>>(
        h_f, c_f, out + (size_t)M * NVOC);
}
